// MultiHeadAttention_1133871366505
// MI455X (gfx1250) — compile-verified, matrix operand order checked by reading, not run on hardware
//
#include <hip/hip_runtime.h>
#include <hip/hip_fp16.h>

typedef __attribute__((ext_vector_type(16))) _Float16 v16h;
typedef __attribute__((ext_vector_type(8)))  _Float16 v8h;
typedef __attribute__((ext_vector_type(8)))  float    v8f;
typedef __attribute__((vector_size(16)))     int      vi4_t;

#define D_MODEL 1024
#define NH      16
#define DKH     64
#define BATCH   2
#define SEQ     2048
#define M_TOTAL (BATCH * SEQ)   // 4096

// ---- gfx1250 async-to-LDS path (guarded; falls back to plain LDS staging) --
#if defined(__HIP_DEVICE_COMPILE__) && defined(__has_builtin)
# if __has_builtin(__builtin_amdgcn_global_load_async_to_lds_b128)
#  define ASYNC_OK 1
# endif
#endif
#ifndef ASYNC_OK
# define ASYNC_OK 0
#endif

template <typename T>
__device__ inline void copy16(T* lds_dst, const T* gsrc) {
#if ASYNC_OK
    __builtin_amdgcn_global_load_async_to_lds_b128(
        (__attribute__((address_space(1))) vi4_t*)(gsrc),
        (__attribute__((address_space(3))) vi4_t*)(lds_dst),
        /*offset=*/0, /*cpol=*/0);
#else
    *(int4*)lds_dst = *(const int4*)gsrc;
#endif
}

template <int N>
__device__ inline void wait_async() {
#if ASYNC_OK
# if __has_builtin(__builtin_amdgcn_s_wait_asynccnt)
    __builtin_amdgcn_s_wait_asynccnt(N);
# else
    asm volatile("s_wait_asynccnt %0" ::"i"(N) : "memory");
# endif
#endif
}

// ---- fragment loaders: 16 contiguous values -> v16h (global or LDS) -------
__device__ inline v16h load16(const float* __restrict__ p) {
    const float4* q = (const float4*)p;
    v16h r;
#pragma unroll
    for (int i = 0; i < 4; ++i) {
        float4 t = q[i];
        r[4 * i + 0] = (_Float16)t.x;
        r[4 * i + 1] = (_Float16)t.y;
        r[4 * i + 2] = (_Float16)t.z;
        r[4 * i + 3] = (_Float16)t.w;
    }
    return r;
}
__device__ inline v16h load16(const _Float16* __restrict__ p) {
    const v8h* q = (const v8h*)p;
    v8h lo = q[0], hi = q[1];
    v16h r;
#pragma unroll
    for (int i = 0; i < 8; ++i) { r[i] = lo[i]; r[8 + i] = hi[i]; }
    return r;
}

__device__ inline v8f wmma_f16(v16h a, v16h b, v8f c) {
    return __builtin_amdgcn_wmma_f32_16x16x32_f16(
        false, a, false, b, (short)0, c, false, false);
}

// ---------------------------------------------------------------------------
// C[M,N] = X[M,K] @ W[N,K]^T + bias   (M=4096, N=K=1024)
// MODE 0: f16 row-major; MODE 1: f16 per-head transposed [B][H][DKH][SEQ];
// MODE 2: f32 row-major.
// Block = 256 threads (8 waves). Block tile 128x128, wave tile 32x64.
// K-step 64: 16 WMMAs per barrier pair. A/B tiles double-buffered in LDS
// via gfx1250 async-to-LDS copies.
// ---------------------------------------------------------------------------
template <typename TIn>
__device__ inline void stage_gemm(TIn* Asm, float* Bsm, const TIn* __restrict__ X,
                                  const float* __restrict__ W, int arow, int nrow,
                                  int kk, int t) {
    constexpr int EA  = 16 / sizeof(TIn);       // elems per 16B seg
    constexpr int SGA = 64 / EA;                // segs per 64-elem row
    constexpr int ITA = 128 * SGA / 256;        // per-thread A segs (f32:8, f16:4)
#pragma unroll
    for (int i = 0; i < ITA; ++i) {
        const int idx = t + i * 256;
        const int row = idx / SGA, sg = idx % SGA;
        copy16(Asm + row * 64 + sg * EA,
               X + (size_t)(arow + row) * D_MODEL + kk + sg * EA);
    }
#pragma unroll
    for (int i = 0; i < 8; ++i) {               // B: 128 rows x 16 segs of 4 floats
        const int idx = t + i * 256;
        const int row = idx >> 4, sg = idx & 15;
        copy16(Bsm + row * 64 + sg * 4,
               W + (size_t)(nrow + row) * D_MODEL + kk + sg * 4);
    }
}

template <typename TIn, int MODE>
__global__ __launch_bounds__(256) void gemm_xwt(
    const TIn* __restrict__ X, const float* __restrict__ W,
    const float* __restrict__ bias, _Float16* __restrict__ out16,
    float* __restrict__ out32) {
    __shared__ __align__(16) TIn   Asm[2][128 * 64];
    __shared__ __align__(16) float Bsm[2][128 * 64];

    const int N = D_MODEL;
    const int t    = threadIdx.x;
    const int lane = t & 31, wave = t >> 5;
    const int l15  = lane & 15, half = lane >> 4;
    const int wrow = wave >> 1, wcol = wave & 1;
    const int arow = blockIdx.y * 128, nrow = blockIdx.x * 128;
    const int mloc = wrow * 32, nloc = wcol * 64;

    constexpr int CNT = (128 * (64 / (16 / (int)sizeof(TIn)))) / 256 + 8;

    stage_gemm(Asm[0], Bsm[0], X, W, arow, nrow, 0, t);

    v8f acc[2][4] = {};
    for (int kk = 0; kk < D_MODEL; kk += 64) {
        const int cur = (kk >> 6) & 1;
        if (kk + 64 < D_MODEL) {
            stage_gemm(Asm[cur ^ 1], Bsm[cur ^ 1], X, W, arow, nrow, kk + 64, t);
            wait_async<CNT>();      // newest batch may fly; chunk kk is done
        } else {
            wait_async<0>();
        }
        if (kk + 128 < D_MODEL)     // prefetch two steps ahead -> global_prefetch_b8
            __builtin_prefetch(X + (size_t)(arow + (t >> 1)) * D_MODEL + kk + 128, 0, 1);
        __syncthreads();

#pragma unroll
        for (int ks = 0; ks < 2; ++ks) {
            const int ksl = ks * 32 + half * 16;
            v16h a[2], b[4];
#pragma unroll
            for (int tm = 0; tm < 2; ++tm)
                a[tm] = load16(&Asm[cur][(mloc + tm * 16 + l15) * 64 + ksl]);
#pragma unroll
            for (int tn = 0; tn < 4; ++tn)
                b[tn] = load16(&Bsm[cur][(nloc + tn * 16 + l15) * 64 + ksl]);
#pragma unroll
            for (int tm = 0; tm < 2; ++tm)
#pragma unroll
                for (int tn = 0; tn < 4; ++tn)
                    acc[tm][tn] = wmma_f16(a[tm], b[tn], acc[tm][tn]);
        }
        __syncthreads();            // buffer reuse fence
    }

    // C/D layout: VGPR j, lanes 0-15 -> (M=j, N=lane); lanes 16-31 -> (M=j+8).
#pragma unroll
    for (int tm = 0; tm < 2; ++tm) {
#pragma unroll
        for (int tn = 0; tn < 4; ++tn) {
            const int n = nrow + nloc + tn * 16 + l15;
            const float bv = bias[n];
#pragma unroll
            for (int j = 0; j < 8; ++j) {
                const int m = arow + mloc + tm * 16 + half * 8 + j;
                const float v = acc[tm][tn][j] + bv;
                if (MODE == 0) {
                    out16[(size_t)m * N + n] = (_Float16)v;
                } else if (MODE == 1) {
                    const int bI = m >> 11, s = m & (SEQ - 1);
                    const int h = n >> 6, d = n & (DKH - 1);
                    out16[(((size_t)(bI * NH + h) * DKH + d) << 11) + s] = (_Float16)v;
                } else {
                    out32[(size_t)m * N + n] = v;
                }
            }
        }
    }
}

// ---------------------------------------------------------------------------
// Flash attention. Grid: (SEQ/128, B*H). Block = 256 (8 waves); each wave owns
// 16 query rows; the block shares double-buffered 64-deep K/V chunks in LDS.
// 16 WMMAs per softmax epoch.
// ---------------------------------------------------------------------------
__device__ inline void stage_kv(_Float16* Ksm, _Float16* Vsm,
                                const _Float16* __restrict__ kbase,
                                const _Float16* __restrict__ vbase, int kv, int t) {
#pragma unroll
    for (int i = 0; i < 2; ++i) {   // K chunk: 64 rows x 64 halfs (8 segs/row)
        const int idx = t + i * 256;
        copy16(Ksm + (idx >> 3) * 64 + (idx & 7) * 8,
               kbase + (size_t)(kv + (idx >> 3)) * D_MODEL + (idx & 7) * 8);
    }
#pragma unroll
    for (int i = 0; i < 2; ++i) {   // V^T chunk: 64 rows(d) x 64 halfs(kv)
        const int idx = t + i * 256;
        copy16(Vsm + (idx >> 3) * 64 + (idx & 7) * 8,
               vbase + (size_t)(idx >> 3) * SEQ + kv + (idx & 7) * 8);
    }
}

__global__ __launch_bounds__(256) void flash_attn(
    const _Float16* __restrict__ qh, const _Float16* __restrict__ kh,
    const _Float16* __restrict__ vT, _Float16* __restrict__ o16) {
    __shared__ __align__(16) _Float16 Ksm[2][64 * 64];
    __shared__ __align__(16) _Float16 Vsm[2][64 * 64];
    __shared__ __align__(64) _Float16 Pl[8][16 * 64];

    const int t    = threadIdx.x;
    const int lane = t & 31, wave = t >> 5;
    const int l15  = lane & 15, half = lane >> 4;
    const int bh = blockIdx.y;
    const int b  = bh >> 4, h = bh & 15;
    const int qrow0 = blockIdx.x * 128 + wave * 16;

    const _Float16* qbase = qh + (size_t)(b * SEQ) * D_MODEL + h * DKH;
    const _Float16* kbase = kh + (size_t)(b * SEQ) * D_MODEL + h * DKH;
    const _Float16* vbase = vT + (size_t)(b * NH + h) * DKH * SEQ;

    v16h aq[2];
#pragma unroll
    for (int s = 0; s < 2; ++s)
        aq[s] = load16(qbase + (size_t)(qrow0 + l15) * D_MODEL + s * 32 + half * 16);

    v8f acc[4] = {};
    v8f mrun, lrun;
#pragma unroll
    for (int j = 0; j < 8; ++j) { mrun[j] = -1e30f; lrun[j] = 0.0f; }

    stage_kv(Ksm[0], Vsm[0], kbase, vbase, 0, t);

    for (int kv = 0; kv < SEQ; kv += 64) {
        const int cur = (kv >> 6) & 1;
        if (kv + 64 < SEQ) {
            stage_kv(Ksm[cur ^ 1], Vsm[cur ^ 1], kbase, vbase, kv + 64, t);
            wait_async<4>();
        } else {
            wait_async<0>();
        }
        __syncthreads();

        // ---- scores S = (Q K^T) / sqrt(dk), 16x64 (4 N-tiles) ----
        v8f sc[4] = {};
#pragma unroll
        for (int s = 0; s < 2; ++s) {
#pragma unroll
            for (int tn = 0; tn < 4; ++tn) {
                v16h kf = load16(&Ksm[cur][(tn * 16 + l15) * 64 + s * 32 + half * 16]);
                sc[tn] = wmma_f16(aq[s], kf, sc[tn]);
            }
        }
        const float scale = 0.125f;
#pragma unroll
        for (int tn = 0; tn < 4; ++tn)
#pragma unroll
            for (int j = 0; j < 8; ++j) sc[tn][j] *= scale;

        // ---- online softmax (row M=j+8*half lives in one 16-lane half) ----
        v8f cmax;
#pragma unroll
        for (int j = 0; j < 8; ++j)
            cmax[j] = fmaxf(fmaxf(sc[0][j], sc[1][j]), fmaxf(sc[2][j], sc[3][j]));
#pragma unroll
        for (int msk = 1; msk < 16; msk <<= 1)
#pragma unroll
            for (int j = 0; j < 8; ++j)
                cmax[j] = fmaxf(cmax[j], __shfl_xor(cmax[j], msk, 32));

        v8f corr;
#pragma unroll
        for (int j = 0; j < 8; ++j) {
            const float mn = fmaxf(mrun[j], cmax[j]);
            corr[j] = __expf(mrun[j] - mn);
            mrun[j] = mn;
        }
        v8f rsum = {};
#pragma unroll
        for (int tn = 0; tn < 4; ++tn)
#pragma unroll
            for (int j = 0; j < 8; ++j) {
                const float p = __expf(sc[tn][j] - mrun[j]);
                sc[tn][j] = p;
                rsum[j] += p;
            }
#pragma unroll
        for (int msk = 1; msk < 16; msk <<= 1)
#pragma unroll
            for (int j = 0; j < 8; ++j) rsum[j] += __shfl_xor(rsum[j], msk, 32);
#pragma unroll
        for (int j = 0; j < 8; ++j) lrun[j] = lrun[j] * corr[j] + rsum[j];
#pragma unroll
        for (int tn = 0; tn < 4; ++tn)
#pragma unroll
            for (int j = 0; j < 8; ++j) acc[tn][j] *= corr[j];

        // ---- C-layout -> A-layout transpose of P (16x64) via per-wave LDS ----
        _Float16* pl = Pl[wave];
#pragma unroll
        for (int tn = 0; tn < 4; ++tn)
#pragma unroll
            for (int j = 0; j < 8; ++j)
                pl[(j + 8 * half) * 64 + tn * 16 + l15] = (_Float16)sc[tn][j];
        v16h pf[2];
#pragma unroll
        for (int s = 0; s < 2; ++s)     // same-wave DS: in-order, no barrier
            pf[s] = load16(pl + l15 * 64 + s * 32 + half * 16);

        // ---- O += P @ V (2 kv-steps x 4 d-tiles) ----
#pragma unroll
        for (int s = 0; s < 2; ++s)
#pragma unroll
            for (int tn = 0; tn < 4; ++tn) {
                v16h vf = load16(&Vsm[cur][(tn * 16 + l15) * 64 + s * 32 + half * 16]);
                acc[tn] = wmma_f16(pf[s], vf, acc[tn]);
            }
        __syncthreads();            // buffer reuse fence
    }

    v8f rl;
#pragma unroll
    for (int j = 0; j < 8; ++j) rl[j] = 1.0f / lrun[j];
    const size_t obase = (size_t)(b * SEQ) * D_MODEL + h * DKH;
#pragma unroll
    for (int tn = 0; tn < 4; ++tn)
#pragma unroll
        for (int j = 0; j < 8; ++j) {
            const int qr = qrow0 + half * 8 + j;
            const int d  = tn * 16 + l15;
            o16[obase + (size_t)qr * D_MODEL + d] = (_Float16)(acc[tn][j] * rl[j]);
        }
}

// ---------------------------------------------------------------------------
extern "C" void kernel_launch(void* const* d_in, const int* in_sizes, int n_in,
                              void* d_out, int out_size, void* d_ws, size_t ws_size,
                              hipStream_t stream) {
    const float* Q  = (const float*)d_in[0];
    const float* K  = (const float*)d_in[1];
    const float* V  = (const float*)d_in[2];
    const float* Wq = (const float*)d_in[3];
    const float* bq = (const float*)d_in[4];
    const float* Wk = (const float*)d_in[5];
    const float* bk = (const float*)d_in[6];
    const float* Wv = (const float*)d_in[7];
    const float* bv = (const float*)d_in[8];
    const float* Wo = (const float*)d_in[9];
    const float* bo = (const float*)d_in[10];
    float* out = (float*)d_out;

    const size_t elems = (size_t)M_TOTAL * D_MODEL;
    _Float16* qws  = (_Float16*)d_ws;
    _Float16* kws  = qws + elems;
    _Float16* vtws = kws + elems;
    _Float16* ows  = vtws + elems;   // 32 MB total f16 scratch

    dim3 blk(256);
    dim3 g1(D_MODEL / 128, M_TOTAL / 128);  // (8, 32)
    gemm_xwt<float, 0><<<g1, blk, 0, stream>>>(Q, Wq, bq, qws, nullptr);
    gemm_xwt<float, 0><<<g1, blk, 0, stream>>>(K, Wk, bk, kws, nullptr);
    gemm_xwt<float, 1><<<g1, blk, 0, stream>>>(V, Wv, bv, vtws, nullptr);

    dim3 g2(SEQ / 128, BATCH * NH);         // (16, 32)
    flash_attn<<<g2, blk, 0, stream>>>(qws, kws, vtws, ows);

    gemm_xwt<_Float16, 2><<<g1, blk, 0, stream>>>(ows, Wo, bo, nullptr, out);
}
